// UnquantizedFusedMoEMethod_46909632807490
// MI455X (gfx1250) — compile-verified, matrix operand order checked by reading, not run on hardware
//
#include <hip/hip_runtime.h>
#include <hip/hip_bf16.h>
#include <stdint.h>

// Problem constants (from reference)
#define T_TOK 4096
#define H_DIM 2048
#define I_DIM 2816
#define E_NUM 8
#define K_TOP 2
#define NPAIR (T_TOK * K_TOP)   // 8192 token-expert pairs
#define M_TILE 32               // tokens per block (2 rowblocks of 16)
#define XS (H_DIM + 8)          // LDS x-tile row stride (bf16 elems), 16B-aligned rows
#define AS (I_DIM + 8)          // LDS act row stride (bf16 elems), 16B-aligned rows
#define XBYTES ((size_t)M_TILE * XS * 2)
#define ABYTES ((size_t)M_TILE * AS * 2)

typedef __attribute__((ext_vector_type(16))) __bf16 bf16x16;
typedef __attribute__((ext_vector_type(8)))  float  f32x8;

union Frag {
    bf16x16  v;
    __bf16   h[16];
    uint32_t u[8];
};

// Raw fp32 fragment: 4 x global_load_b128, no converts attached, so multiple
// fragments' loads can be claused and kept in flight simultaneously.
struct RawFrag { float4 a, b, c, d; };

// CDNA5 16-bit fragment layout: lane<16 holds K {0..7,16..23}, lane>=16 holds
// K {8..15,24..31}; kb=(lane<16)?0:8. Elements kb..kb+7 and kb+16..kb+23 are
// contiguous -> four b128 loads.
__device__ __forceinline__ RawFrag load_raw(const float* __restrict__ p, int kb) {
    RawFrag r;
    r.a = *(const float4*)(p + kb);
    r.b = *(const float4*)(p + kb + 4);
    r.c = *(const float4*)(p + kb + 16);
    r.d = *(const float4*)(p + kb + 20);
    return r;
}

// fp32 -> bf16 fragment conversion (lowers to v_cvt_pk_bf16_f32 pairs).
__device__ __forceinline__ void cvt_frag(const RawFrag& r, Frag& f) {
    f.h[0]  = (__bf16)r.a.x; f.h[1]  = (__bf16)r.a.y; f.h[2]  = (__bf16)r.a.z; f.h[3]  = (__bf16)r.a.w;
    f.h[4]  = (__bf16)r.b.x; f.h[5]  = (__bf16)r.b.y; f.h[6]  = (__bf16)r.b.z; f.h[7]  = (__bf16)r.b.w;
    f.h[8]  = (__bf16)r.c.x; f.h[9]  = (__bf16)r.c.y; f.h[10] = (__bf16)r.c.z; f.h[11] = (__bf16)r.c.w;
    f.h[12] = (__bf16)r.d.x; f.h[13] = (__bf16)r.d.y; f.h[14] = (__bf16)r.d.z; f.h[15] = (__bf16)r.d.w;
}

// Same fragment pattern from bf16 data already staged in LDS: two ds_load_b128.
__device__ __forceinline__ void load_frag_lds(const uint16_t* p, int kb, Frag& f) {
    *(uint4*)&f.u[0] = *(const uint4*)(p + kb);
    *(uint4*)&f.u[4] = *(const uint4*)(p + kb + 16);
}

__device__ __forceinline__ uint16_t bf16bits(float a) {
    union { __bf16 b; uint16_t s; } c;
    c.b = (__bf16)a;
    return c.s;
}

__global__ void zero_out_kernel(float* __restrict__ out) {
    size_t i = (size_t)blockIdx.x * blockDim.x + threadIdx.x;
    if (i < (size_t)T_TOK * H_DIM) out[i] = 0.0f;
}

__global__ void route_init_kernel(int* __restrict__ cnt) {
    if (threadIdx.x < E_NUM) cnt[threadIdx.x] = 0;
}

__global__ void route_kernel(const int* __restrict__ topk_ids,
                             const float* __restrict__ topk_w,
                             int* __restrict__ cnt,
                             int* __restrict__ tok,
                             float* __restrict__ tw) {
    int i = blockIdx.x * blockDim.x + threadIdx.x;   // over T*K
    if (i < NPAIR) {
        int e = topk_ids[i];
        int pos = atomicAdd(&cnt[e], 1);
        tok[e * NPAIR + pos] = i / K_TOP;
        tw[e * NPAIR + pos]  = topk_w[i];
    }
}

// LDS (305 KB) pins occupancy to 1 workgroup/WGP = 2 waves/SIMD. Tell the
// compiler so it budgets VGPRs for 2 waves/EU instead of default occupancy:
// this frees registers for deep load pipelining around the WMMAs.
__global__ void __launch_bounds__(256)
__attribute__((amdgpu_waves_per_eu(2)))
moe_fused_kernel(const float* __restrict__ x,
                 const float* __restrict__ w13,
                 const float* __restrict__ w2,
                 const int* __restrict__ cnt,
                 const int* __restrict__ tokList,
                 const float* __restrict__ twList,
                 float* __restrict__ out) {
    extern __shared__ char smem[];
    uint16_t* xLds   = (uint16_t*)smem;                      // M_TILE x XS bf16 (x tile)
    uint16_t* actLds = (uint16_t*)(smem + XBYTES);           // M_TILE x AS bf16 (act tile)
    int*      tokLds = (int*)(smem + XBYTES + ABYTES);
    float*    twLds  = (float*)(tokLds + M_TILE);

    const int e    = blockIdx.y;
    const int n_e  = cnt[e];
    const int base = blockIdx.x * M_TILE;
    if (base >= n_e) return;                    // uniform exit: EXEC stays full

    const int lane = threadIdx.x & 31;
    const int wc   = threadIdx.x >> 5;          // 0..7: column group (1x8 wave layout)
    const int kb   = (lane < 16) ? 0 : 8;
    const int ln   = lane & 15;
    const int mhi  = (lane < 16) ? 0 : 8;

    if (threadIdx.x < M_TILE) {
        int idx  = base + (int)threadIdx.x;
        int idxc = (idx < n_e) ? idx : (n_e - 1);       // clamp: valid token, weight 0
        tokLds[threadIdx.x] = tokList[e * NPAIR + idxc];
        twLds[threadIdx.x]  = (idx < n_e) ? twList[e * NPAIR + idxc] : 0.0f;
    }
    __syncthreads();

    // ---- Stage x tile to LDS as bf16: each element loaded+converted ONCE per block ----
    for (int idx = threadIdx.x; idx < M_TILE * (H_DIM / 2); idx += blockDim.x) {
        int r  = idx >> 10;                     // idx / 1024
        int c2 = idx & (H_DIM / 2 - 1);         // pair index within row
        const float2 v = *(const float2*)(x + (size_t)tokLds[r] * H_DIM + 2 * c2);
        uint32_t pk = (uint32_t)bf16bits(v.x) | ((uint32_t)bf16bits(v.y) << 16);
        *(uint32_t*)&xLds[r * XS + 2 * c2] = pk;
    }
    __syncthreads();

    const float*    wgBase = w13 + (size_t)e * (2 * I_DIM) * H_DIM;
    const uint16_t* aX0    = xLds + (size_t)(ln)      * XS;   // rowblock 0 (tokens 0-15)
    const uint16_t* aX1    = xLds + (size_t)(16 + ln) * XS;   // rowblock 1 (tokens 16-31)

    // ---- Phase 1: act[32 x I] = silu(x@Wg^T) * (x@Wu^T), bf16 into LDS ----
    // Each wave owns 16 columns but BOTH rowblocks: every B fragment (weight row)
    // is loaded+converted once and feeds 2 WMMAs. Unroll 2 k-steps so two sets of
    // loads can be in flight while the matrix pipe runs.
    for (int ib = wc * 16; ib < I_DIM; ib += 128) {
        f32x8 accG0 = {}, accG1 = {}, accU0 = {}, accU1 = {};
        const float* gRow = wgBase + (size_t)(ib + ln) * H_DIM;           // gate row i
        const float* uRow = wgBase + (size_t)(I_DIM + ib + ln) * H_DIM;   // up row i
#pragma unroll 2
        for (int h = 0; h < H_DIM; h += 32) {
            RawFrag rg = load_raw(gRow + h, kb);    // 4x global_load_b128
            RawFrag ru = load_raw(uRow + h, kb);    // 4x global_load_b128
            Frag fa0, fa1, fg, fu;
            load_frag_lds(aX0 + h, kb, fa0);        // 2x ds_load_b128
            load_frag_lds(aX1 + h, kb, fa1);        // 2x ds_load_b128
            cvt_frag(rg, fg);
            cvt_frag(ru, fu);
            accG0 = __builtin_amdgcn_wmma_f32_16x16x32_bf16(
                false, fa0.v, false, fg.v, (short)0, accG0, false, false);
            accG1 = __builtin_amdgcn_wmma_f32_16x16x32_bf16(
                false, fa1.v, false, fg.v, (short)0, accG1, false, false);
            accU0 = __builtin_amdgcn_wmma_f32_16x16x32_bf16(
                false, fa0.v, false, fu.v, (short)0, accU0, false, false);
            accU1 = __builtin_amdgcn_wmma_f32_16x16x32_bf16(
                false, fa1.v, false, fu.v, (short)0, accU1, false, false);
        }
#pragma unroll
        for (int v = 0; v < 8; ++v) {                   // D layout: M = v + mhi, N = ln
            float g0 = accG0[v], u0 = accU0[v];
            float g1 = accG1[v], u1 = accU1[v];
            float a0 = g0 * __builtin_amdgcn_rcpf(1.0f + __expf(-g0)) * u0;
            float a1 = g1 * __builtin_amdgcn_rcpf(1.0f + __expf(-g1)) * u1;
            actLds[(v + mhi)      * AS + (ib + ln)] = bf16bits(a0);
            actLds[(16 + v + mhi) * AS + (ib + ln)] = bf16bits(a1);
        }
    }
    __syncthreads();

    // ---- Phase 2: y[32 x H] = act @ W2[e]^T, scale by routing weight, atomic add ----
    const float*    w2Base = w2 + (size_t)e * H_DIM * I_DIM;
    const uint16_t* aP0    = actLds + (size_t)(ln)      * AS;
    const uint16_t* aP1    = actLds + (size_t)(16 + ln) * AS;
    for (int hb = wc * 16; hb < H_DIM; hb += 128) {
        f32x8 acc0 = {}, acc1 = {};
        const float* bRow = w2Base + (size_t)(hb + ln) * I_DIM;           // w2 row h
#pragma unroll 2
        for (int i = 0; i < I_DIM; i += 32) {
            RawFrag rb = load_raw(bRow + i, kb);    // 4x global_load_b128
            Frag fa0, fa1, fb;
            load_frag_lds(aP0 + i, kb, fa0);
            load_frag_lds(aP1 + i, kb, fa1);
            cvt_frag(rb, fb);
            acc0 = __builtin_amdgcn_wmma_f32_16x16x32_bf16(
                false, fa0.v, false, fb.v, (short)0, acc0, false, false);
            acc1 = __builtin_amdgcn_wmma_f32_16x16x32_bf16(
                false, fa1.v, false, fb.v, (short)0, acc1, false, false);
        }
#pragma unroll
        for (int v = 0; v < 8; ++v) {
            int   m0 = v + mhi;
            int   m1 = 16 + v + mhi;
            float w0 = twLds[m0];
            float w1 = twLds[m1];
            if (w0 != 0.0f)
                atomicAdd(out + (size_t)tokLds[m0] * H_DIM + (hb + ln), acc0[v] * w0);
            if (w1 != 0.0f)
                atomicAdd(out + (size_t)tokLds[m1] * H_DIM + (hb + ln), acc1[v] * w1);
        }
    }
}

extern "C" void kernel_launch(void* const* d_in, const int* in_sizes, int n_in,
                              void* d_out, int out_size, void* d_ws, size_t ws_size,
                              hipStream_t stream) {
    (void)in_sizes; (void)n_in; (void)out_size; (void)ws_size;
    const float* x   = (const float*)d_in[0];
    const float* w13 = (const float*)d_in[1];
    const float* w2  = (const float*)d_in[2];
    const float* tkw = (const float*)d_in[3];
    const int*   tki = (const int*)d_in[4];    // per harness: integer -> const int*
    float* out = (float*)d_out;

    // workspace layout: [cnt: E ints][tok: E*NPAIR ints][tw: E*NPAIR floats] ~520 KB
    char*  ws  = (char*)d_ws;
    int*   cnt = (int*)ws;
    int*   tok = (int*)(ws + 256);
    float* tw  = (float*)(ws + 256 + (size_t)E_NUM * NPAIR * sizeof(int));

    route_init_kernel<<<1, 64, 0, stream>>>(cnt);
    route_kernel<<<(NPAIR + 255) / 256, 256, 0, stream>>>(tki, tkw, cnt, tok, tw);

    size_t nOut = (size_t)T_TOK * H_DIM;
    zero_out_kernel<<<(int)((nOut + 255) / 256), 256, 0, stream>>>(out);

    // grid.y = expert (expert-major residency -> one expert's 69MB weight set lives in 192MB L2)
    // grid.x = worst-case tiles per expert; empty tiles exit immediately.
    dim3 grid(NPAIR / M_TILE, E_NUM);
    size_t shm = XBYTES + ABYTES + (size_t)M_TILE * (sizeof(int) + sizeof(float));
    moe_fused_kernel<<<grid, 256, shm, stream>>>(x, w13, w2, cnt, tok, tw, out);
}
